// DecoderLayer_60954175865249
// MI455X (gfx1250) — matrix operand order repaired and checked by reading, not compiled, not run on hardware
//
#include <hip/hip_runtime.h>

// ---------------------------------------------------------------------------
// CDNA5 (gfx1250, wave32) decoder layer.
// bf16 operands (weights pre-transposed), v_wmma_f32_16x16x32_bf16 compute,
// TDM (tensor_load_to_lds) double-buffered LDS staging with s_wait_tensorcnt.
// ---------------------------------------------------------------------------

typedef __attribute__((ext_vector_type(16))) __bf16 v16bf;
typedef __attribute__((ext_vector_type(8)))  __bf16 v8bf;
typedef __attribute__((ext_vector_type(8)))  float  v8f;
typedef unsigned int u32x4 __attribute__((ext_vector_type(4)));
typedef int          i32x4 __attribute__((ext_vector_type(4)));
typedef int          i32x8 __attribute__((ext_vector_type(8)));

#define LAS 40  // LDS row stride (bf16): 64B data + 16B TDM pad -> conflict-free frags
#define LBS 40

// ---------------------------------------------------------------------------
// TDM: DMA one 2D bf16 tile (rows x 32 elems, row stride = stride_elems) into
// LDS at lds_addr, padding 16B after every 64B row (LDS row stride 80B).
// Descriptor layout per CDNA5 ISA ch.8 (D# groups 0/1; groups 2/3 zero = 2D).
// ---------------------------------------------------------------------------
__device__ __forceinline__ void tdm_load_2d(unsigned lds_addr, const void* gptr,
                                            unsigned rows, unsigned stride_elems) {
    const unsigned long long ga = (unsigned long long)(size_t)gptr;
    u32x4 g0 = { 1u,                                   // count=1 valid descriptor
                 lds_addr,                             // LDS byte address
                 (unsigned)ga,                         // global_addr[31:0]
                 ((unsigned)(ga >> 32) & 0x01ffffffu)  // global_addr[56:32]
                     | 0x80000000u };                  // type=2 ("image")
    const unsigned w0 = (1u << 16)    // data_size = 2 bytes
                      | (1u << 20)    // pad_enable
                      | (3u << 22)    // pad_interval: every 16 DWORDs (64B row)
                      | (3u << 25);   // pad_amount: 4 DWORDs (16B)
    const unsigned td0 = stride_elems;   // tensor_dim0 >= tile width
    const unsigned td1 = 1u << 20;       // tensor_dim1: large (all rows in-bounds)
    i32x8 g1 = { (int)w0,
                 (int)(td0 << 16),                      // tensor_dim0[15:0]
                 (int)((td0 >> 16) | (td1 << 16)),      // td0[31:16] | td1[15:0]
                 (int)((td1 >> 16) | (32u << 16)),      // td1[31:16] | tile_dim0=32
                 (int)(rows & 0xffffu),                 // tile_dim1 (tile_dim2=0)
                 (int)stride_elems,                     // tensor_dim0_stride[31:0]
                 0, 0 };
    i32x4 z4 = { 0, 0, 0, 0 };
#if defined(__clang_major__) && (__clang_major__ >= 23)
    i32x8 z8 = { 0, 0, 0, 0, 0, 0, 0, 0 };
    __builtin_amdgcn_tensor_load_to_lds(g0, g1, z4, z4, z8, 0);
#else
    __builtin_amdgcn_tensor_load_to_lds(g0, g1, z4, z4, 0);
#endif
}

struct GemmP {
    const __bf16* A;   // [M,K], leading dim lda
    const __bf16* B;   // [N,K], leading dim ldb (K-contiguous "T layout")
    float*  Cf;        // optional fp32 out [M,N] ldc
    __bf16* Cb;        // optional bf16 out [M,N] ldc
    __bf16* Ct;        // optional bf16 transposed out [N,M] ldt (batch-1 users)
    const float* bias; // [N] or null
    const float* res;  // fp32 residual added into Cf, or null
    int M, N, K, lda, ldb, ldc, ldt, H;
    float scale;
    long long sAb, sAh, sBb, sBh, sCb, sCh;  // batch offsets: (z/H)*sb + (z%H)*sh
};

template <bool RELU>
__global__ __launch_bounds__(256) void gemm_wmma(GemmP p) {
    // block tile 128(M) x 64(N); 8 waves in 4(M) x 2(N); wave tile 32x32.
    __shared__ alignas(16) __bf16 sA[2][128 * LAS];
    __shared__ alignas(16) __bf16 sB[2][64 * LBS];

    const int tid  = threadIdx.x;
    const int lane = tid & 31;
    const int wave = tid >> 5;
    const int m0 = blockIdx.y << 7;
    const int n0 = blockIdx.x << 6;
    const int zb = (int)blockIdx.z / p.H;
    const int zh = (int)blockIdx.z % p.H;

    const __bf16* Ag = p.A + (size_t)zb * p.sAb + (size_t)zh * p.sAh;
    const __bf16* Bg = p.B + (size_t)zb * p.sBb + (size_t)zh * p.sBh;

    const int wm = (wave & 3) << 5;   // 0,32,64,96
    const int wn = (wave >> 2) << 5;  // 0,32

    v8f acc00 = {}, acc01 = {}, acc10 = {}, acc11 = {};

    // TDM staging: wave 0 drives the DMA for the whole block.
    const unsigned ldsA[2] = { (unsigned)(size_t)&sA[0][0], (unsigned)(size_t)&sA[1][0] };
    const unsigned ldsB[2] = { (unsigned)(size_t)&sB[0][0], (unsigned)(size_t)&sB[1][0] };
    const __bf16* Abase = Ag + (size_t)m0 * p.lda;
    const __bf16* Bbase = Bg + (size_t)n0 * p.ldb;

    if (wave == 0) {
        tdm_load_2d(ldsA[0], Abase, 128, (unsigned)p.lda);
        tdm_load_2d(ldsB[0], Bbase, 64,  (unsigned)p.ldb);
    }

    int cur = 0;
    for (int kk = 0; kk < p.K; kk += 32) {
        if (wave == 0) {
            if (kk + 32 < p.K) {
                // queue next tile pair into the other buffer, then wait for the
                // previous pair (TDM completes in order per wave: cnt<=2 => ready)
                tdm_load_2d(ldsA[cur ^ 1], Abase + (kk + 32), 128, (unsigned)p.lda);
                tdm_load_2d(ldsB[cur ^ 1], Bbase + (kk + 32), 64,  (unsigned)p.ldb);
                __builtin_amdgcn_s_wait_tensorcnt((short)2);
            } else {
                __builtin_amdgcn_s_wait_tensorcnt((short)0);
            }
        }
        __syncthreads();   // buf[cur] is DMA-complete for all waves

        // ---- fragments (ISA wave32 layouts) ----
        // A 16x32: lanes 0-15 K {0..7,16..23}; lanes 16-31 K {8..15,24..31}
        const int akb = (lane < 16) ? 0 : 8;
        const __bf16* a0p = &sA[cur][(wm + (lane & 15)) * LAS + akb];
        const __bf16* a1p = a0p + 16 * LAS;
        v16bf a0 = __builtin_shufflevector(*(const v8bf*)a0p, *(const v8bf*)(a0p + 16),
                       0,1,2,3,4,5,6,7,8,9,10,11,12,13,14,15);
        v16bf a1 = __builtin_shufflevector(*(const v8bf*)a1p, *(const v8bf*)(a1p + 16),
                       0,1,2,3,4,5,6,7,8,9,10,11,12,13,14,15);
        // B 32x16: lane = N col; lanes 0-15 K {0..7,16..23}; lanes 16-31 K {8..15,24..31}
        const int bkb = (lane < 16) ? 0 : 8;
        const __bf16* b0p = &sB[cur][(wn + (lane & 15)) * LBS + bkb];
        const __bf16* b1p = b0p + 16 * LBS;
        v16bf b0 = __builtin_shufflevector(*(const v8bf*)b0p, *(const v8bf*)(b0p + 16),
                       0,1,2,3,4,5,6,7,8,9,10,11,12,13,14,15);
        v16bf b1 = __builtin_shufflevector(*(const v8bf*)b1p, *(const v8bf*)(b1p + 16),
                       0,1,2,3,4,5,6,7,8,9,10,11,12,13,14,15);

        acc00 = __builtin_amdgcn_wmma_f32_16x16x32_bf16(false, a0, false, b0, (short)0, acc00, false, false);
        acc01 = __builtin_amdgcn_wmma_f32_16x16x32_bf16(false, a0, false, b1, (short)0, acc01, false, false);
        acc10 = __builtin_amdgcn_wmma_f32_16x16x32_bf16(false, a1, false, b0, (short)0, acc10, false, false);
        acc11 = __builtin_amdgcn_wmma_f32_16x16x32_bf16(false, a1, false, b1, (short)0, acc11, false, false);

        __syncthreads();   // reads of buf[cur] done before TDM refills it
        cur ^= 1;
    }

    // ---- epilogue: C/D 16x16 f32 layout: VGPR r -> row r (lanes 0-15) / r+8 ----
    float*  Cf = p.Cf ? p.Cf + (size_t)zb * p.sCb + (size_t)zh * p.sCh : nullptr;
    __bf16* Cb = p.Cb ? p.Cb + (size_t)zb * p.sCb + (size_t)zh * p.sCh : nullptr;
    __bf16* Ct = p.Ct;
    const int colb = lane & 15;
    const int rsel = (lane < 16) ? 0 : 8;
#pragma unroll
    for (int ti = 0; ti < 2; ++ti) {
#pragma unroll
        for (int tj = 0; tj < 2; ++tj) {
            const v8f acc = ti ? (tj ? acc11 : acc10) : (tj ? acc01 : acc00);
            const int col   = n0 + wn + (tj << 4) + colb;
            const int rbase = m0 + wm + (ti << 4) + rsel;
            const float bv = p.bias ? p.bias[col] : 0.0f;
            float vv[8];
#pragma unroll
            for (int r = 0; r < 8; ++r) {
                float t = acc[r] * p.scale + bv;
                if (RELU) t = fmaxf(t, 0.0f);
                vv[r] = t;
            }
            if (Cf) {
#pragma unroll
                for (int r = 0; r < 8; ++r) {
                    const size_t idx = (size_t)(rbase + r) * p.ldc + col;
                    Cf[idx] = p.res ? vv[r] + p.res[idx] : vv[r];
                }
            }
            if (Cb) {
#pragma unroll
                for (int r = 0; r < 8; ++r)
                    Cb[(size_t)(rbase + r) * p.ldc + col] = (__bf16)vv[r];
            }
            if (Ct) {  // 8 consecutive rows -> one contiguous b128 store transposed
                union { __bf16 h[8]; uint4 u; } pk;
#pragma unroll
                for (int r = 0; r < 8; ++r) pk.h[r] = (__bf16)vv[r];
                *(uint4*)&Ct[(size_t)col * p.ldt + rbase] = pk.u;
            }
        }
    }
}

// ---------------------------------------------------------------------------
// fp32 -> bf16 conversion passes (run once per call; weights also transposed)
// ---------------------------------------------------------------------------
__global__ __launch_bounds__(256) void cvt_bf16(const float* __restrict__ in,
                                                __bf16* __restrict__ out, int n) {
    const int i = (blockIdx.x * 256 + threadIdx.x) * 4;
    if (i + 3 < n) {
        float4 f = *(const float4*)(in + i);
        union { __bf16 h[4]; uint2 u; } pk;
        pk.h[0] = (__bf16)f.x; pk.h[1] = (__bf16)f.y;
        pk.h[2] = (__bf16)f.z; pk.h[3] = (__bf16)f.w;
        *(uint2*)(out + i) = pk.u;
    }
}

// in: fp32 [R,C] -> out: bf16 [C,R]
__global__ __launch_bounds__(256) void cvt_bf16_t(const float* __restrict__ in,
                                                  __bf16* __restrict__ out, int R, int C) {
    __shared__ float t[32][33];
    const int c0 = blockIdx.x << 5, r0 = blockIdx.y << 5;
    const int tx = threadIdx.x & 31, ty = threadIdx.x >> 5;
#pragma unroll
    for (int i = 0; i < 4; ++i)
        t[ty + i * 8][tx] = in[(size_t)(r0 + ty + i * 8) * C + c0 + tx];
    __syncthreads();
#pragma unroll
    for (int i = 0; i < 4; ++i)
        out[(size_t)(c0 + ty + i * 8) * R + r0 + tx] = (__bf16)t[tx][ty + i * 8];
}

// ---------------------------------------------------------------------------
// Block reductions (wave32 shuffles + LDS across 8 waves; blockDim == 256)
// ---------------------------------------------------------------------------
__device__ inline float block_sum(float v, float* red) {
#pragma unroll
    for (int o = 16; o > 0; o >>= 1) v += __shfl_xor(v, o, 32);
    if ((threadIdx.x & 31) == 0) red[threadIdx.x >> 5] = v;
    __syncthreads();
    float r = 0.0f;
#pragma unroll
    for (int i = 0; i < 8; ++i) r += red[i];
    __syncthreads();
    return r;
}

__device__ inline float block_max(float v, float* red) {
#pragma unroll
    for (int o = 16; o > 0; o >>= 1) v = fmaxf(v, __shfl_xor(v, o, 32));
    if ((threadIdx.x & 31) == 0) red[threadIdx.x >> 5] = v;
    __syncthreads();
    float r = red[0];
#pragma unroll
    for (int i = 1; i < 8; ++i) r = fmaxf(r, red[i]);
    __syncthreads();
    return r;
}

// outF/outB = (res?) + LN(x)*g + b   (either output optional)
__global__ __launch_bounds__(256) void ln_kernel(
    const float* __restrict__ x, const float* __restrict__ res,
    const float* __restrict__ g, const float* __restrict__ b,
    float* __restrict__ outF, __bf16* __restrict__ outB, int d) {
    __shared__ float red[8];
    const int tid = threadIdx.x;
    const size_t base = (size_t)blockIdx.x * d;
    const int E = d >> 8;
    float vals[4];
    float s = 0.0f, q = 0.0f;
    for (int e = 0; e < E; ++e) {
        float v = x[base + tid + (e << 8)];
        vals[e] = v; s += v; q += v * v;
    }
    const float invd = 1.0f / (float)d;
    const float mean = block_sum(s, red) * invd;
    const float var  = block_sum(q, red) * invd - mean * mean;
    const float inv  = rsqrtf(var + 1e-5f);
    for (int e = 0; e < E; ++e) {
        const int j = tid + (e << 8);
        float o = (vals[e] - mean) * inv * g[j] + b[j];
        if (res) o += res[base + j];
        if (outF) outF[base + j] = o;
        if (outB) outB[base + j] = (__bf16)o;
    }
}

// masked softmax; optional fp32 writeback + optional bf16 probability copy
__global__ __launch_bounds__(256) void softmax_mask(
    float* __restrict__ sc, const int* __restrict__ mask, __bf16* __restrict__ pb,
    int L, int rows_per_b, int writeF32) {
    __shared__ float red[8];
    const int row = blockIdx.x;
    const int bidx = row / rows_per_b;
    float* rp = sc + (size_t)row * L;
    const int* mp = mask + (size_t)bidx * L;
    const int tid = threadIdx.x;
    const int E = L >> 8;
    float vals[4];
    float mx = -3.0e38f;
    for (int e = 0; e < E; ++e) {
        const int j = tid + (e << 8);
        float v = rp[j];
        if (mp[j] == 0) v = -65500.0f;
        vals[e] = v;
        mx = fmaxf(mx, v);
    }
    const float bm = block_max(mx, red);
    float s = 0.0f;
    for (int e = 0; e < E; ++e) { vals[e] = __expf(vals[e] - bm); s += vals[e]; }
    const float invs = 1.0f / block_sum(s, red);
    for (int e = 0; e < E; ++e) {
        const int j = tid + (e << 8);
        const float pv = vals[e] * invs;
        if (writeF32) rp[j] = pv;
        if (pb) pb[(size_t)row * L + j] = (__bf16)pv;
    }
}

// ---------------------------------------------------------------------------
// Host-side launch helpers
// ---------------------------------------------------------------------------
static inline void run_gemm(hipStream_t st, bool relu,
                            const __bf16* A, const __bf16* Bm,
                            float* Cf, __bf16* Cb, __bf16* Ct,
                            const float* bias, const float* res,
                            int M, int N, int K, int lda, int ldb, int ldc, int ldt,
                            float scale, int batch, int H,
                            long long sAb, long long sAh,
                            long long sBb, long long sBh,
                            long long sCb, long long sCh) {
    GemmP p;
    p.A = A; p.B = Bm; p.Cf = Cf; p.Cb = Cb; p.Ct = Ct; p.bias = bias; p.res = res;
    p.M = M; p.N = N; p.K = K; p.lda = lda; p.ldb = ldb; p.ldc = ldc; p.ldt = ldt;
    p.H = H; p.scale = scale;
    p.sAb = sAb; p.sAh = sAh; p.sBb = sBb; p.sBh = sBh; p.sCb = sCb; p.sCh = sCh;
    dim3 g(N / 64, M / 128, batch), blk(256);
    if (relu) gemm_wmma<true ><<<g, blk, 0, st>>>(p);
    else      gemm_wmma<false><<<g, blk, 0, st>>>(p);
}

extern "C" void kernel_launch(void* const* d_in, const int* in_sizes, int n_in,
                              void* d_out, int out_size, void* d_ws, size_t ws_size,
                              hipStream_t stream) {
    (void)in_sizes; (void)n_in; (void)out_size; (void)ws_size;

    const int B = 8, T = 512, S = 1024, D = 512, DFF = 2048, H = 8;
    const int MT = B * T;   // 4096
    const int MS = B * S;   // 8192
    const float iscale = 0.125f;  // 1/sqrt(64)

    const float* x     = (const float*)d_in[0];
    const float* feat  = (const float*)d_in[1];
    const int*   smask = (const int*)d_in[5];
    const int*   tmask = (const int*)d_in[6];
    const float* W[10] = { (const float*)d_in[7],  (const float*)d_in[9],
                           (const float*)d_in[11], (const float*)d_in[13],
                           (const float*)d_in[15], (const float*)d_in[17],
                           (const float*)d_in[19], (const float*)d_in[21],
                           (const float*)d_in[23], (const float*)d_in[25] };
    const float* sa_q_b = (const float*)d_in[8];  const float* sa_k_b = (const float*)d_in[10];
    const float* sa_v_b = (const float*)d_in[12]; const float* sa_o_b = (const float*)d_in[14];
    const float* ca_q_b = (const float*)d_in[16]; const float* ca_k_b = (const float*)d_in[18];
    const float* ca_v_b = (const float*)d_in[20]; const float* ca_o_b = (const float*)d_in[22];
    const float* f_b1 = (const float*)d_in[24];   const float* f_b2 = (const float*)d_in[26];
    const float* n1g = (const float*)d_in[27]; const float* n1b = (const float*)d_in[28];
    const float* n2g = (const float*)d_in[29]; const float* n2b = (const float*)d_in[30];
    const float* n3g = (const float*)d_in[31]; const float* n3b = (const float*)d_in[32];

    // ---- workspace carve-up ----
    char* base = (char*)d_ws;
    size_t off = 0;
    auto allocF = [&](size_t n) { off = (off + 255) & ~(size_t)255; float* p = (float*)(base + off); off += n * 4; return p; };
    auto allocH = [&](size_t n) { off = (off + 255) & ~(size_t)255; __bf16* p = (__bf16*)(base + off); off += n * 2; return p; };
    const size_t MTD = (size_t)MT * D, MSD = (size_t)MS * D;

    float*  scores = allocF((size_t)B * H * T * T);   // 64MB; dead after self-softmax
    __bf16* p2bf   = (__bf16*)scores;                 // alias (cross P bf16 copy)
    __bf16* f1bf   = (__bf16*)scores;                 // alias (FFN hidden)
    __bf16* xbf    = allocH(MTD);
    __bf16* featbf = allocH(MSD);
    __bf16* WT[10];
    for (int i = 0; i < 8; ++i) WT[i] = allocH((size_t)D * D);
    WT[8] = allocH((size_t)D * DFF);
    WT[9] = allocH((size_t)DFF * D);
    __bf16* qbf   = allocH(MTD);
    __bf16* kbf   = allocH(MTD);
    __bf16* vbfT  = allocH(MTD);                      // [D][MT] transposed
    __bf16* pbf   = allocH((size_t)B * H * T * T);
    __bf16* attbf = allocH(MTD);
    float*  att2  = allocF(MTD);
    float*  x1f   = allocF(MTD);
    __bf16* x1bf  = allocH(MTD);
    __bf16* k2bf  = allocH(MSD);
    __bf16* v2bfT = allocH(MSD);                      // [D][MS] transposed
    __bf16* cattbf= allocH(MTD);
    float*  srco  = allocF(MTD);
    float*  x2    = allocF(MTD);
    __bf16* hbbf  = allocH(MTD);

    float* out_x = (float*)d_out;
    float* out_p = out_x + MTD;   // [B,H,T,S] cross-attn probabilities

    // ---- one-time conversions: activations to bf16, weights to bf16 transposed ----
    cvt_bf16<<<dim3((int)(MTD / 1024)), dim3(256), 0, stream>>>(x, xbf, (int)MTD);
    cvt_bf16<<<dim3((int)(MSD / 1024)), dim3(256), 0, stream>>>(feat, featbf, (int)MSD);
    for (int i = 0; i < 8; ++i)
        cvt_bf16_t<<<dim3(D / 32, D / 32), dim3(256), 0, stream>>>(W[i], WT[i], D, D);
    cvt_bf16_t<<<dim3(DFF / 32, D / 32), dim3(256), 0, stream>>>(W[8], WT[8], D, DFF);
    cvt_bf16_t<<<dim3(D / 32, DFF / 32), dim3(256), 0, stream>>>(W[9], WT[9], DFF, D);

    // ---------------- self-attention ----------------
    run_gemm(stream, false, xbf, WT[0], nullptr, qbf,  nullptr, sa_q_b, nullptr,
             MT, D, D, D, D, D, 0, 1.0f, 1, 1, 0,0,0,0,0,0);
    run_gemm(stream, false, xbf, WT[1], nullptr, kbf,  nullptr, sa_k_b, nullptr,
             MT, D, D, D, D, D, 0, 1.0f, 1, 1, 0,0,0,0,0,0);
    run_gemm(stream, false, xbf, WT[2], nullptr, nullptr, vbfT, sa_v_b, nullptr,
             MT, D, D, D, D, D, MT, 1.0f, 1, 1, 0,0,0,0,0,0);
    // scores[b,h] = scale * Qh @ Kh^T
    run_gemm(stream, false, qbf, kbf, scores, nullptr, nullptr, nullptr, nullptr,
             T, T, 64, D, D, T, 0, iscale, B * H, H,
             (long long)T * D, 64, (long long)T * D, 64,
             (long long)H * T * T, (long long)T * T);
    softmax_mask<<<dim3(B * H * T), dim3(256), 0, stream>>>(scores, tmask, pbf, T, H * T, 0);
    // att[b,:,h*64:] = P @ Vh   (V supplied transposed: [D][MT])
    run_gemm(stream, false, pbf, vbfT, nullptr, attbf, nullptr, nullptr, nullptr,
             T, 64, T, T, MT, D, 0, 1.0f, B * H, H,
             (long long)H * T * T, (long long)T * T,
             (long long)T, (long long)64 * MT,
             (long long)T * D, 64);
    run_gemm(stream, false, attbf, WT[3], att2, nullptr, nullptr, sa_o_b, nullptr,
             MT, D, D, D, D, D, 0, 1.0f, 1, 1, 0,0,0,0,0,0);
    ln_kernel<<<dim3(MT), dim3(256), 0, stream>>>(att2, x, n1g, n1b, x1f, x1bf, D);

    // ---------------- cross-attention ----------------
    run_gemm(stream, false, x1bf, WT[4], nullptr, qbf, nullptr, ca_q_b, nullptr,
             MT, D, D, D, D, D, 0, 1.0f, 1, 1, 0,0,0,0,0,0);
    run_gemm(stream, false, featbf, WT[5], nullptr, k2bf, nullptr, ca_k_b, nullptr,
             MS, D, D, D, D, D, 0, 1.0f, 1, 1, 0,0,0,0,0,0);
    run_gemm(stream, false, featbf, WT[6], nullptr, nullptr, v2bfT, ca_v_b, nullptr,
             MS, D, D, D, D, D, MS, 1.0f, 1, 1, 0,0,0,0,0,0);
    run_gemm(stream, false, qbf, k2bf, out_p, nullptr, nullptr, nullptr, nullptr,
             T, S, 64, D, D, S, 0, iscale, B * H, H,
             (long long)T * D, 64, (long long)S * D, 64,
             (long long)H * T * S, (long long)T * S);
    softmax_mask<<<dim3(B * H * T), dim3(256), 0, stream>>>(out_p, smask, p2bf, S, H * T, 1);
    run_gemm(stream, false, p2bf, v2bfT, nullptr, cattbf, nullptr, nullptr, nullptr,
             T, 64, S, S, MS, D, 0, 1.0f, B * H, H,
             (long long)H * T * S, (long long)T * S,
             (long long)S, (long long)64 * MS,
             (long long)T * D, 64);
    run_gemm(stream, false, cattbf, WT[7], srco, nullptr, nullptr, ca_o_b, nullptr,
             MT, D, D, D, D, D, 0, 1.0f, 1, 1, 0,0,0,0,0,0);
    ln_kernel<<<dim3(MT), dim3(256), 0, stream>>>(srco, x1f, n2g, n2b, x2, nullptr, D);

    // ---------------- pre-norm FFN ----------------
    ln_kernel<<<dim3(MT), dim3(256), 0, stream>>>(x2, nullptr, n3g, n3b, nullptr, hbbf, D);
    run_gemm(stream, true, hbbf, WT[8], nullptr, f1bf, nullptr, f_b1, nullptr,
             MT, DFF, D, D, D, DFF, 0, 1.0f, 1, 1, 0,0,0,0,0,0);
    run_gemm(stream, false, f1bf, WT[9], out_x, nullptr, nullptr, f_b2, x2,
             MT, D, DFF, DFF, DFF, D, 0, 1.0f, 1, 1, 0,0,0,0,0,0);
}
